// VectorQuantizerEMA_17643725652360
// MI455X (gfx1250) — matrix operand order repaired and checked by reading, not compiled, not run on hardware
//
#include <hip/hip_runtime.h>
#include <hip/hip_bf16.h>

// ---- CDNA5 vector types -----------------------------------------------------
typedef __bf16 v16bf __attribute__((ext_vector_type(16)));
typedef float  v8f   __attribute__((ext_vector_type(8)));
typedef float  f4    __attribute__((ext_vector_type(4)));
// 4-byte-aligned float4 for stores into d_out (region starts at d_out+1 -> 4B misaligned)
typedef float  f4u   __attribute__((ext_vector_type(4), aligned(4)));

#define EDIM   256      // embedding dim
#define KCODES 1024     // num embeddings
#define NROWS  65536    // 1024*64 flattened rows
#define ROWS_PER_BLOCK 128
#define NWAVES 8
#define NTILES (KCODES / 16)               // 64
#define NBLOCKS (NROWS / ROWS_PER_BLOCK)   // 512

// Workgroup barrier that fences ONLY LDS (ds) traffic. The default
// __syncthreads() flushes loadcnt to 0 as well, which would force our
// in-flight global prefetch of the next codebook tile to land before the
// barrier. CDNA5 split counters let us wait on dscnt alone.
__device__ __forceinline__ void lds_barrier() {
  asm volatile(
      "s_wait_dscnt 0x0\n\t"
      "s_barrier_signal -1\n\t"
      "s_barrier_wait -1" ::: "memory");
}

// ---------------------------------------------------------------------------
// Kernel 1: codebook squared norms ||e_k||^2  -> ws[0..1023]
// ---------------------------------------------------------------------------
__global__ __launch_bounds__(256) void vq_bnorm_kernel(
    const float* __restrict__ codebook, float* __restrict__ bnorm) {
  const int lane = threadIdx.x & 31;
  const int w    = threadIdx.x >> 5;
  const int code = blockIdx.x * NWAVES + w;
  const float* p = codebook + (size_t)code * EDIM + lane * 8;
  f4 a = *(const f4*)p;
  f4 b = *(const f4*)(p + 4);
  float s = a[0]*a[0] + a[1]*a[1] + a[2]*a[2] + a[3]*a[3]
          + b[0]*b[0] + b[1]*b[1] + b[2]*b[2] + b[3]*b[3];
#pragma unroll
  for (int off = 1; off < 32; off <<= 1) s += __shfl_xor(s, off, 32);
  if (lane == 0) bnorm[code] = s;
}

// ---------------------------------------------------------------------------
// Kernel 2: fused GEMM(argmin) + outputs.
// Each wave owns 16 input rows, scans all 1024 codes in 16-wide tiles.
// dot(x,e) via bf16 hi/lo split: hi*hi + hi*lo + lo*hi  (3 WMMAs / k-chunk).
// Codebook tiles double-buffered in LDS; the next tile's global fetch and the
// next bnorm value are issued before a dscnt-only barrier so their latency
// hides behind the 24 WMMAs.
// ---------------------------------------------------------------------------
__global__ __launch_bounds__(256) void vq_main_kernel(
    const float* __restrict__ inputs, const float* __restrict__ codebook,
    const float* __restrict__ bnorm, float* __restrict__ outQ,
    float* __restrict__ outOH, float* __restrict__ outIdx,
    float* __restrict__ wsPartial) {
  __shared__ __align__(64) __bf16 sBhi[2][16 * EDIM];
  __shared__ __align__(64) __bf16 sBlo[2][16 * EDIM];
  __shared__ int   sIdx[ROWS_PER_BLOCK];
  __shared__ float sLoss[NWAVES];

  const int tid  = threadIdx.x;
  const int lane = tid & 31;
  const int w    = tid >> 5;
  const int rowBase  = blockIdx.x * ROWS_PER_BLOCK;
  const int waveRow0 = rowBase + w * 16;

  // ---- load this wave's A tile (16 rows x 256) as bf16 hi/lo in VGPRs ----
  // A layout: lane<16 -> row=lane, K 0..15 contiguous; lane>=16 -> K 16..31
  const int arow = waveRow0 + (lane & 15);
  const int koff = (lane >> 4) * 16;
  v16bf Ahi[8], Alo[8];
#pragma unroll
  for (int c = 0; c < 8; ++c) {
    const float* p = inputs + (size_t)arow * EDIM + c * 32 + koff;
    f4 x0 = *(const f4*)(p);
    f4 x1 = *(const f4*)(p + 4);
    f4 x2 = *(const f4*)(p + 8);
    f4 x3 = *(const f4*)(p + 12);
    float xs[16];
#pragma unroll
    for (int j = 0; j < 4; ++j) {
      xs[j] = x0[j]; xs[4 + j] = x1[j]; xs[8 + j] = x2[j]; xs[12 + j] = x3[j];
    }
#pragma unroll
    for (int e = 0; e < 16; ++e) {
      float  f = xs[e];
      __bf16 h = (__bf16)f;
      Ahi[c][e] = h;
      Alo[c][e] = (__bf16)(f - (float)h);
    }
  }

  float minv[8];
  int   mini[8];
#pragma unroll
  for (int r = 0; r < 8; ++r) { minv[r] = 3.4e38f; mini[r] = 0; }

  // B operand addressing: lane<16 -> code=lane, K 0..15 contiguous; lane>=16 -> K 16..31
  const int boff = (lane & 15) * EDIM + (lane >> 4) * 16;

  // ---- software-pipeline prologue: fetch tile 0 + bnorm(0) into registers ----
  f4 ra, rb, rc, rd;
  {
    const float* src = codebook + tid * 16;
    ra = *(const f4*)(src);
    rb = *(const f4*)(src + 4);
    rc = *(const f4*)(src + 8);
    rd = *(const f4*)(src + 12);
  }
  float bnCur = bnorm[lane & 15];
  float bnNext = bnCur;

  for (int t = 0; t < NTILES; ++t) {
    const int buf = t & 1;
    {  // convert + store tile t (held in registers) into LDS buffer
      float xs[16];
#pragma unroll
      for (int j = 0; j < 4; ++j) {
        xs[j] = ra[j]; xs[4 + j] = rb[j]; xs[8 + j] = rc[j]; xs[12 + j] = rd[j];
      }
      const int base = tid * 16;
#pragma unroll
      for (int e = 0; e < 16; ++e) {
        float  f = xs[e];
        __bf16 h = (__bf16)f;
        sBhi[buf][base + e] = h;
        sBlo[buf][base + e] = (__bf16)(f - (float)h);
      }
    }
    // issue global fetch of tile t+1 now; latency hides behind the WMMAs below
    if (t + 1 < NTILES) {
      const float* src = codebook + (size_t)(t + 1) * 16 * EDIM + tid * 16;
      ra = *(const f4*)(src);
      rb = *(const f4*)(src + 4);
      rc = *(const f4*)(src + 8);
      rd = *(const f4*)(src + 12);
      bnNext = bnorm[(t + 1) * 16 + (lane & 15)];
    }
    lds_barrier();  // dscnt-only fence: keeps the prefetch loads in flight

    v8f acc = {0.f, 0.f, 0.f, 0.f, 0.f, 0.f, 0.f, 0.f};
#pragma unroll
    for (int c = 0; c < 8; ++c) {
      v16bf bhi = *(const v16bf*)(&sBhi[buf][boff + c * 32]);
      v16bf blo = *(const v16bf*)(&sBlo[buf][boff + c * 32]);
      acc = __builtin_amdgcn_wmma_f32_16x16x32_bf16(false, Ahi[c], false, bhi,
                                                    (short)0, acc, false, false);
      acc = __builtin_amdgcn_wmma_f32_16x16x32_bf16(false, Ahi[c], false, blo,
                                                    (short)0, acc, false, false);
      acc = __builtin_amdgcn_wmma_f32_16x16x32_bf16(false, Alo[c], false, bhi,
                                                    (short)0, acc, false, false);
    }
    const int cand = t * 16 + (lane & 15);
#pragma unroll
    for (int r = 0; r < 8; ++r) {
      float s = bnCur - 2.0f * acc[r];  // ||e||^2 - 2 x.e (||x||^2 const per row)
      if (s < minv[r]) { minv[r] = s; mini[r] = cand; }
    }
    bnCur = bnNext;
  }

  // ---- argmin across the 16 lanes of each half (n = lane&15) ----
#pragma unroll
  for (int r = 0; r < 8; ++r) {
    float v = minv[r];
    int   i = mini[r];
#pragma unroll
    for (int off = 1; off < 16; off <<= 1) {
      float ov = __shfl_xor(v, off, 32);
      int   oi = __shfl_xor(i, off, 32);
      if (ov < v || (ov == v && oi < i)) { v = ov; i = oi; }
    }
    minv[r] = v; mini[r] = i;
  }
  if ((lane & 15) == 0) {
    const int half = lane >> 4;  // rows: half0 -> m=r, half1 -> m=r+8
#pragma unroll
    for (int r = 0; r < 8; ++r) sIdx[w * 16 + half * 8 + r] = mini[r];
  }
  __syncthreads();

  // ---- indices output (as float; reference returns (S*N,1) ints) ----
  if (tid < ROWS_PER_BLOCK) outIdx[rowBase + tid] = (float)sIdx[tid];

  // ---- quantized gather + exact f32 loss partial ----
  float lp = 0.f;
  for (int m = 0; m < 16; ++m) {
    const int row = waveRow0 + m;
    const int idx = sIdx[w * 16 + m];
    const float* cb = codebook + (size_t)idx * EDIM + lane * 8;
    const float* xr = inputs + (size_t)row * EDIM + lane * 8;
    f4 q0 = *(const f4*)cb, q1 = *(const f4*)(cb + 4);
    f4 x0 = *(const f4*)xr, x1 = *(const f4*)(xr + 4);
    f4u* dst = (f4u*)(outQ + (size_t)row * EDIM + lane * 8);
    f4u t0 = q0, t1 = q1;
    __builtin_nontemporal_store(t0, dst);
    __builtin_nontemporal_store(t1, dst + 1);
    f4 d0 = q0 - x0, d1 = q1 - x1;
    lp += d0[0]*d0[0] + d0[1]*d0[1] + d0[2]*d0[2] + d0[3]*d0[3]
        + d1[0]*d1[0] + d1[1]*d1[1] + d1[2]*d1[2] + d1[3]*d1[3];
  }
#pragma unroll
  for (int off = 1; off < 32; off <<= 1) lp += __shfl_xor(lp, off, 32);
  if (lane == 0) sLoss[w] = lp;

  // ---- one-hot rows (non-temporal; 268MB > L2) ----
  for (int m = 0; m < 16; ++m) {
    const int row = waveRow0 + m;
    const int idx = sIdx[w * 16 + m];
    f4u* dst = (f4u*)(outOH + (size_t)row * KCODES);
    const int qq = idx >> 2, cc = idx & 3;
#pragma unroll
    for (int j = 0; j < 8; ++j) {
      const int quad = lane + 32 * j;
      f4u v; v[0] = 0.f; v[1] = 0.f; v[2] = 0.f; v[3] = 0.f;
      if (quad == qq) v[cc] = 1.0f;
      __builtin_nontemporal_store(v, dst + quad);
    }
  }

  __syncthreads();
  if (tid == 0) {
    float s = 0.f;
#pragma unroll
    for (int i = 0; i < NWAVES; ++i) s += sLoss[i];
    wsPartial[blockIdx.x] = s;  // deterministic per-block partial
  }
}

// ---------------------------------------------------------------------------
// Kernel 3: deterministic final loss reduction
// ---------------------------------------------------------------------------
__global__ void vq_loss_kernel(const float* __restrict__ parts,
                               float* __restrict__ outLoss) {
  if (threadIdx.x == 0 && blockIdx.x == 0) {
    float s = 0.f;
    for (int i = 0; i < NBLOCKS; ++i) s += parts[i];
    // loss = COMMITMENT_COST * mean((q - x)^2), mean over 65536*256 elements
    outLoss[0] = 0.25f * s / 16777216.0f;
  }
}

extern "C" void kernel_launch(void* const* d_in, const int* in_sizes, int n_in,
                              void* d_out, int out_size, void* d_ws, size_t ws_size,
                              hipStream_t stream) {
  (void)in_sizes; (void)n_in; (void)out_size; (void)ws_size;
  const float* inputs   = (const float*)d_in[0];  // (1024,64,256) f32
  const float* codebook = (const float*)d_in[1];  // (1024,256)   f32

  float* out     = (float*)d_out;
  float* outLoss = out;                                   // 1
  float* outQ    = out + 1;                               // 16777216
  float* outOH   = out + 1 + 16777216;                    // 67108864
  float* outIdx  = out + 1 + 16777216 + 67108864;         // 65536

  float* wsBnorm   = (float*)d_ws;      // 1024 floats
  float* wsPartial = wsBnorm + 1024;    // 512 floats

  vq_bnorm_kernel<<<KCODES / NWAVES, 256, 0, stream>>>(codebook, wsBnorm);
  vq_main_kernel<<<NBLOCKS, 256, 0, stream>>>(inputs, codebook, wsBnorm, outQ,
                                              outOH, outIdx, wsPartial);
  vq_loss_kernel<<<1, 1, 0, stream>>>(wsPartial, outLoss);
}
